// MultiHeadSelfAttention_51659866636827
// MI455X (gfx1250) — matrix operand order repaired and checked by reading, not compiled, not run on hardware
//
#include <hip/hip_runtime.h>
#include <hip/hip_bf16.h>

// ---------------- problem constants (match reference) ----------------
#define B_      4
#define S_      2048
#define H_      16
#define DH_     64
#define DM_     1024
#define INDIM_  1024
#define MT_     (B_ * S_)          // 8192 total rows (b*S + s)
#define NEGV    (-9.0e15f)

typedef __attribute__((ext_vector_type(16))) __bf16 bf16x16;
typedef __attribute__((ext_vector_type(8)))  __bf16 bf16x8;
typedef __attribute__((ext_vector_type(8)))  float  f32x8;

// D = A(16x32 bf16) * B(32x16 bf16) + C(16x16 f32), wave32 WMMA
__device__ __forceinline__ f32x8 wmma_bf16(bf16x16 a, bf16x16 b, f32x8 c) {
  return __builtin_amdgcn_wmma_f32_16x16x32_bf16(
      /*neg_a=*/false, a, /*neg_b=*/false, b,
      /*c_mod=*/(short)0, c, /*reuse_a=*/false, /*reuse_b=*/false);
}

__device__ __forceinline__ bf16x16 cat8(bf16x8 lo, bf16x8 hi) {
  return __builtin_shufflevector(lo, hi, 0,1,2,3,4,5,6,7,8,9,10,11,12,13,14,15);
}

// A-operand fragment: row-major A, lane m=lane&15, kh=lane>>4
//  elems 0..7  = A[m][k0 + 8*kh + 0..7]
//  elems 8..15 = A[m][k0 + 16 + 8*kh + 0..7]
__device__ __forceinline__ bf16x16 load_a_frag(const __bf16* base, int ld, int k0) {
  const int lane = threadIdx.x & 31;
  const int m = lane & 15, kh = lane >> 4;
  const __bf16* p = base + (size_t)m * ld + k0 + kh * 8;
  bf16x8 lo = *(const bf16x8*)p;
  bf16x8 hi = *(const bf16x8*)(p + 16);
  return cat8(lo, hi);
}

// B-operand fragment: B[k][n] = Wrows[n][k] (Wrows row-major, K contiguous)
//  lane n=lane&15, kh=lane>>4: elems 0..7 = Wrows[n][k0 + 8*kh + 0..7], elems 8..15 = Wrows[n][k0 + 16 + 8*kh + 0..7]
__device__ __forceinline__ bf16x16 load_b_frag(const __bf16* rowbase, int ld, int k0) {
  const int lane = threadIdx.x & 31;
  const int n = lane & 15, kh = lane >> 4;
  const __bf16* p = rowbase + (size_t)n * ld + k0 + kh * 8;
  return cat8(*(const bf16x8*)p, *(const bf16x8*)(p + 16));
}

// ---------------- fp32 -> bf16 cast ----------------
__global__ void cvt_f32_bf16(const float* __restrict__ src, __bf16* __restrict__ dst, int n) {
  for (int i = blockIdx.x * blockDim.x + threadIdx.x; i < n; i += gridDim.x * blockDim.x)
    dst[i] = (__bf16)src[i];
}

// ---------------- generic GEMM: y[M,N] = A[M,K] @ W[N,K]^T + bias ----------------
// 32x64 output tile per wave: 8 WMMAs per K=32 step (2 A frags x 4 B frags).
// mode 0: write bf16 at [b, h, s, d]        (Q / K layout)
// mode 1: write bf16 at [b, h, d, s]        (V transposed for P@V B-operand)
// mode 2: write f32  at [m, n] (d_out)      (output projection)
__global__ void gemm_wmma_bf16(const __bf16* __restrict__ A,
                               const __bf16* __restrict__ W,
                               const float*  __restrict__ bias,
                               void* __restrict__ out,
                               int M, int N, int K, int mode) {
  const int wave   = (blockIdx.x * blockDim.x + threadIdx.x) >> 5;
  const int ntiles = N >> 6;                 // 64-wide N tiles per wave
  const int mt = wave / ntiles;
  const int nt = wave % ntiles;
  if (mt >= (M >> 5)) return;                // uniform per wave
  const int row0 = mt << 5;                  // 32 rows
  const int col0 = nt << 6;                  // 64 cols
  const int lane = threadIdx.x & 31;
  const int ln = lane & 15, lh = lane >> 4;

  f32x8 acc[2][4] = { { {}, {}, {}, {} }, { {}, {}, {}, {} } };
  const __bf16* Arow0 = A + (size_t)row0 * K;
  const __bf16* Arow1 = Arow0 + (size_t)16 * K;
  const __bf16* Wrow  = W + (size_t)col0 * K;

  for (int k0 = 0; k0 < K; k0 += 32) {
    bf16x16 a0 = load_a_frag(Arow0, K, k0);
    bf16x16 a1 = load_a_frag(Arow1, K, k0);
    if (k0 + 32 < K) {   // hint next K-slice of A into cache (global_prefetch_b8)
      __builtin_prefetch(Arow0 + (size_t)(lane & 15) * K + k0 + 32, 0, 1);
      __builtin_prefetch(Arow1 + (size_t)(lane & 15) * K + k0 + 32, 0, 1);
    }
#pragma unroll
    for (int t = 0; t < 4; ++t) {
      bf16x16 b = load_b_frag(Wrow + (size_t)(16 * t) * K, K, k0);
      acc[0][t] = wmma_bf16(a0, b, acc[0][t]);
      acc[1][t] = wmma_bf16(a1, b, acc[1][t]);
    }
  }

  // C layout: VGPR r -> row = r + 8*lh, col = 16*t + ln
#pragma unroll
  for (int mi = 0; mi < 2; ++mi) {
#pragma unroll
    for (int t = 0; t < 4; ++t) {
#pragma unroll
      for (int r = 0; r < 8; ++r) {
        const int m   = row0 + 16 * mi + r + 8 * lh;
        const int col = col0 + 16 * t + ln;
        const float v = acc[mi][t][r] + bias[col];
        if (mode == 2) {
          ((float*)out)[(size_t)m * N + col] = v;
        } else {
          const int h = col >> 6, d = col & 63;
          const int bb = m / S_, s = m % S_;
          size_t idx;
          if (mode == 0) idx = (((size_t)bb * H_ + h) * S_ + s) * DH_ + d;
          else           idx = (((size_t)bb * H_ + h) * DH_ + d) * S_ + s;
          ((__bf16*)out)[idx] = (__bf16)v;
        }
      }
    }
  }
}

// ---------------- causal flash attention, one wave per (b, h, 16-query tile) ----------------
// Q,K: [B,H,S,64] bf16.  Vt: [B,H,64,S] bf16.  O: [B,S,1024] bf16 (heads merged).
__global__ void attn_wmma(const __bf16* __restrict__ Q,
                          const __bf16* __restrict__ Km,
                          const __bf16* __restrict__ Vt,
                          __bf16* __restrict__ O) {
  __shared__ __bf16 lds_p[8][16][32];        // per-wave 16x32 P tile (bf16), 8 waves/block

  const int wid  = threadIdx.x >> 5;
  const int wave = (blockIdx.x * blockDim.x + threadIdx.x) >> 5;
  const int qt = wave & ((S_ >> 4) - 1);     // 128 query tiles per (b,h)
  const int bh = wave >> 7;
  if (bh >= B_ * H_) return;
  const int q0 = qt << 4;
  const int lane = threadIdx.x & 31;
  const int ln = lane & 15, lh = lane >> 4;

  const __bf16* qbase = Q  + ((size_t)bh * S_ + q0) * DH_;
  const __bf16* kbase = Km + (size_t)bh * S_ * DH_;
  const __bf16* vbase = Vt + (size_t)bh * DH_ * S_;

  // resident Q A-fragments (head dim split K=0..31 / 32..63)
  const bf16x16 qa0 = load_a_frag(qbase, DH_, 0);
  const bf16x16 qa1 = load_a_frag(qbase, DH_, 32);

  float m_run[8], l_run[8];
#pragma unroll
  for (int r = 0; r < 8; ++r) { m_run[r] = -3.0e38f; l_run[r] = 0.f; }
  f32x8 o[4] = { {}, {}, {}, {} };

  const int nchunks = (q0 + 47) >> 5;        // keys 0 .. q0+15, 32 per chunk
  for (int c = 0; c < nchunks; ++c) {
    const int key0 = c << 5;

    // scores: two 16(q)x16(k) subtiles, K-dim = head dim (2 x K=32 WMMA)
    f32x8 st0 = {}, st1 = {};
    {
      const __bf16* k0p = kbase + (size_t)key0 * DH_;
      st0 = wmma_bf16(qa0, load_b_frag(k0p, DH_, 0),  st0);
      st0 = wmma_bf16(qa1, load_b_frag(k0p, DH_, 32), st0);
      const __bf16* k1p = k0p + (size_t)16 * DH_;
      st1 = wmma_bf16(qa0, load_b_frag(k1p, DH_, 0),  st1);
      st1 = wmma_bf16(qa1, load_b_frag(k1p, DH_, 32), st1);
    }

    // scale (1/sqrt(64)), causal mask, online softmax. Row r+8*lh lives across
    // the 16-lane half-wave -> shfl_xor{1,2,4,8} row reductions.
    float p0[8], p1[8], alpha[8];
#pragma unroll
    for (int r = 0; r < 8; ++r) {
      const int qrow = q0 + r + 8 * lh;
      float s0 = st0[r] * 0.125f;
      float s1 = st1[r] * 0.125f;
      s0 = (key0 + ln      > qrow) ? NEGV : s0;
      s1 = (key0 + 16 + ln > qrow) ? NEGV : s1;
      float mx = fmaxf(s0, s1);
      mx = fmaxf(mx, __shfl_xor(mx, 1, 32));
      mx = fmaxf(mx, __shfl_xor(mx, 2, 32));
      mx = fmaxf(mx, __shfl_xor(mx, 4, 32));
      mx = fmaxf(mx, __shfl_xor(mx, 8, 32));
      const float mnew = fmaxf(m_run[r], mx);
      const float a = __expf(m_run[r] - mnew);
      p0[r] = __expf(s0 - mnew);
      p1[r] = __expf(s1 - mnew);
      float rs = p0[r] + p1[r];
      rs += __shfl_xor(rs, 1, 32);
      rs += __shfl_xor(rs, 2, 32);
      rs += __shfl_xor(rs, 4, 32);
      rs += __shfl_xor(rs, 8, 32);
      l_run[r] = l_run[r] * a + rs;
      m_run[r] = mnew;
      alpha[r] = a;
    }
#pragma unroll
    for (int t = 0; t < 4; ++t)
#pragma unroll
      for (int r = 0; r < 8; ++r)
        o[t][r] *= alpha[r];

    // P (C layout) -> LDS -> A-operand layout
#pragma unroll
    for (int r = 0; r < 8; ++r) {
      lds_p[wid][r + 8 * lh][ln]      = (__bf16)p0[r];
      lds_p[wid][r + 8 * lh][16 + ln] = (__bf16)p1[r];
    }
    asm volatile("s_wait_dscnt 0" ::: "memory");   // same-wave DS RAW
    bf16x16 pA;
    {
      const __bf16* lp = &lds_p[wid][ln][lh * 8];
      bf16x8 lo = *(const bf16x8*)lp;
      bf16x8 hi = *(const bf16x8*)(lp + 16);
      pA = cat8(lo, hi);
    }

    // o[16x64] += P[16x32] @ V[32x64]; Vt row-major in s => contiguous B loads
#pragma unroll
    for (int t = 0; t < 4; ++t) {
      bf16x16 vb = load_b_frag(vbase + (size_t)(16 * t) * S_, S_, key0);
      o[t] = wmma_bf16(pA, vb, o[t]);
    }
  }

  // epilogue: o / l, merge heads into [B,S,1024] bf16
  const int b = bh / H_, h = bh % H_;
#pragma unroll
  for (int r = 0; r < 8; ++r) {
    const float inv = 1.0f / l_run[r];
    const size_t m = (size_t)b * S_ + q0 + r + 8 * lh;
#pragma unroll
    for (int t = 0; t < 4; ++t)
      O[m * DM_ + (size_t)h * DH_ + 16 * t + ln] = (__bf16)(o[t][r] * inv);
  }
}

// ---------------- host launch ----------------
extern "C" void kernel_launch(void* const* d_in, const int* in_sizes, int n_in,
                              void* d_out, int out_size, void* d_ws, size_t ws_size,
                              hipStream_t stream) {
  const float* x  = (const float*)d_in[0];
  const float* Wq = (const float*)d_in[1];
  const float* bq = (const float*)d_in[2];
  const float* Wk = (const float*)d_in[3];
  const float* bk = (const float*)d_in[4];
  const float* Wv = (const float*)d_in[5];
  const float* bv = (const float*)d_in[6];
  const float* Wo = (const float*)d_in[7];
  const float* bo = (const float*)d_in[8];

  // workspace carve-up (bf16 elements), total ~88 MB
  __bf16* p   = (__bf16*)d_ws;
  __bf16* xb  = p; p += (size_t)MT_ * INDIM_;
  __bf16* wqb = p; p += (size_t)DM_ * INDIM_;
  __bf16* wkb = p; p += (size_t)DM_ * INDIM_;
  __bf16* wvb = p; p += (size_t)DM_ * INDIM_;
  __bf16* wob = p; p += (size_t)DM_ * DM_;
  __bf16* qb  = p; p += (size_t)MT_ * DM_;
  __bf16* kb  = p; p += (size_t)MT_ * DM_;
  __bf16* vtb = p; p += (size_t)MT_ * DM_;
  __bf16* aob = p;

  dim3 blk(256);
  cvt_f32_bf16<<<4096, blk, 0, stream>>>(x,  xb,  MT_ * INDIM_);
  cvt_f32_bf16<<<1024, blk, 0, stream>>>(Wq, wqb, DM_ * INDIM_);
  cvt_f32_bf16<<<1024, blk, 0, stream>>>(Wk, wkb, DM_ * INDIM_);
  cvt_f32_bf16<<<1024, blk, 0, stream>>>(Wv, wvb, DM_ * INDIM_);
  cvt_f32_bf16<<<1024, blk, 0, stream>>>(Wo, wob, DM_ * DM_);

  const int gemm_waves  = (MT_ / 32) * (DM_ / 64);   // 4096 waves (32x64 tiles)
  const int gemm_blocks = gemm_waves * 32 / 256;     // 512 blocks of 8 waves

  gemm_wmma_bf16<<<gemm_blocks, blk, 0, stream>>>(xb, wqb, bq, qb,  MT_, DM_, INDIM_, 0);
  gemm_wmma_bf16<<<gemm_blocks, blk, 0, stream>>>(xb, wkb, bk, kb,  MT_, DM_, INDIM_, 0);
  gemm_wmma_bf16<<<gemm_blocks, blk, 0, stream>>>(xb, wvb, bv, vtb, MT_, DM_, INDIM_, 1);

  const int attn_waves  = B_ * H_ * (S_ / 16);       // 8192 waves
  const int attn_blocks = attn_waves * 32 / 256;     // 1024 blocks
  attn_wmma<<<attn_blocks, blk, 0, stream>>>(qb, kb, vtb, aob);

  gemm_wmma_bf16<<<gemm_blocks, blk, 0, stream>>>(aob, wob, bo, d_out, MT_, DM_, DM_, 2);
}
